// PairwiseScore_49254684950905
// MI455X (gfx1250) — matrix operand order repaired and checked by reading, not compiled, not run on hardware
//
#include <hip/hip_runtime.h>
#include <hip/hip_bf16.h>
#include <stdint.h>

// ---------------------------------------------------------------------------
// Types for CDNA5 WMMA (wave32): V_WMMA_F32_16X16X32_BF16
// ---------------------------------------------------------------------------
typedef __attribute__((ext_vector_type(16))) __bf16 v16bf;
typedef __attribute__((ext_vector_type(8)))  float  v8f;

union Frag {
    uint4 u[2];
    v16bf v;
};

// Problem constants (sizes also read from in_sizes at launch where variable)
#define GDIM   1000
#define GIJ_K  3060
#define KP     3072          // GIJ padded to multiple of 32
#define HDIM   150
#define HP     160           // H padded to multiple of 16
#define NT1    10            // HP/16 N-tiles
#define KS1    96            // KP/32 K-steps for GEMM1
#define KS2    5             // HP/32 K-steps for GEMM2
#define CHUNK  768           // K elements staged in LDS per chunk
#define KS_PER_CHUNK 24      // CHUNK/32
#define NCHUNKS 4            // KP/CHUNK
#define ROWS   16            // pairs per workgroup (M-tile)
#define APAD   8             // bf16 pad per LDS row (bank-conflict avoidance)
#define KMAXP1 129           // K_MAX + 1 (grid row width)

// ---------------------------------------------------------------------------
// Pack an fp32 weight matrix (K x N, row major) into bf16 B-fragment layout,
// zero-padded to (nks*32) x (NT1*16).  Per ISA 16-bit B 32x16 layout:
// lane l holds, for element j in 0..15:  k = kstep*32 + (j>>3)*16 + (l>>4)*8 + (j&7),
//                                        n = ntile*16 + (l&15).
// Flat index: ((kstep*NT1 + ntile)*32 + lane)*16 + j  -> each lane's fragment
// is 32 contiguous bytes (two b128 loads).
// ---------------------------------------------------------------------------
__global__ __launch_bounds__(256)
void pack_w_bf16(const float* __restrict__ W, __bf16* __restrict__ Wp,
                 int K, int N, int total) {
    int idx = blockIdx.x * 256 + threadIdx.x;
    if (idx >= total) return;
    int j    = idx & 15;
    int lane = (idx >> 4) & 31;
    int blk  = idx >> 9;             // kstep*NT1 + ntile
    int nt   = blk % NT1;
    int ks   = blk / NT1;
    int k    = ks * 32 + ((j >> 3) << 4) + ((lane >> 4) << 3) + (j & 7);
    int n    = nt * 16 + (lane & 15);
    float v  = (k < K && n < N) ? W[(size_t)k * N + n] : 0.0f;
    Wp[idx]  = (__bf16)v;
}

// ---------------------------------------------------------------------------
// Output grid init: everything = 1000.0, element [0,0] = 1.0
// ---------------------------------------------------------------------------
__global__ __launch_bounds__(256)
void init_grid(float* __restrict__ grid, int n) {
    int i = blockIdx.x * 256 + threadIdx.x;
    if (i < n) grid[i] = (i == 0) ? 1.0f : 1000.0f;
}

// ---------------------------------------------------------------------------
// Exclusive prefix-sum of group lengths -> starts (M small: single block)
// ---------------------------------------------------------------------------
__global__ __launch_bounds__(256)
void scan_starts(const int* __restrict__ lengths, int* __restrict__ starts, int M) {
    __shared__ int part[256];
    int t = threadIdx.x;
    int per = (M + 255) >> 8;
    int base = t * per;
    int s = 0;
    for (int i = 0; i < per; ++i) { int g = base + i; if (g < M) s += lengths[g]; }
    part[t] = s;
    __syncthreads();
    if (t == 0) {
        int a = 0;
        for (int i = 0; i < 256; ++i) { int v = part[i]; part[i] = a; a += v; }
    }
    __syncthreads();
    int a = part[t];
    for (int i = 0; i < per; ++i) {
        int g = base + i;
        if (g < M) { starts[g] = a; a += lengths[g]; }
    }
}

// ---------------------------------------------------------------------------
// Main fused kernel: 16 pairs per workgroup, 64 threads = 2 waves.
//   - stage bf16 feature tile [16 x 768] in LDS (4 chunks cover K=3072)
//   - GEMM1: each wave 24 ksteps x 5 ntiles of wmma_f32_16x16x32_bf16
//   - bias1 + ReLU -> bf16 h-tile in LDS (A-fragment addressable)
//   - GEMM2: 5 ksteps x 5 ntiles per wave
//   - bias2 + ReLU + dot(W3) via lane shuffles -> per-pair score
// ---------------------------------------------------------------------------
__global__ __launch_bounds__(64)
void pair_mlp(const float* __restrict__ g_i,
              const float* __restrict__ mention_scores,
              const float* __restrict__ dist_emb,
              const float* __restrict__ genre_emb,
              const float* __restrict__ speaker_emb,
              const __bf16* __restrict__ W1b,
              const float* __restrict__ b1,
              const __bf16* __restrict__ W2b,
              const float* __restrict__ b2,
              const float* __restrict__ W3,
              const float* __restrict__ b3,
              const int* __restrict__ mention_ids,
              const int* __restrict__ antecedent_ids,
              const int* __restrict__ dist_ids,
              const int* __restrict__ genre_ids,
              const int* __restrict__ speaker_ids,
              float* __restrict__ scores, int P) {
    __shared__ __align__(16) __bf16 Atile[ROWS][CHUNK + APAD];  // 24.8 KB
    __shared__ __align__(16) __bf16 Htile[ROWS][HP + APAD];     // 5.4 KB
    __shared__ int   mi_s[ROWS], ai_s[ROWS], di_s[ROWS], ge_s[ROWS], sp_s[ROWS];
    __shared__ float red[ROWS][2];

    const int t    = threadIdx.x;
    const int wave = t >> 5;
    const int lane = t & 31;
    const int half = lane >> 4;
    const int mrow = lane & 15;
    const int p0   = blockIdx.x * ROWS;

    if (t < ROWS) {
        int p = p0 + t;
        int ok = (p < P);
        mi_s[t] = ok ? mention_ids[p]    : 0;
        ai_s[t] = ok ? antecedent_ids[p] : 0;
        di_s[t] = ok ? dist_ids[p]       : 0;
        ge_s[t] = ok ? genre_ids[p]      : 0;
        sp_s[t] = ok ? speaker_ids[p]    : 0;
    }
    __syncthreads();

    v8f acc[5];
    const v8f vzero = {0.f, 0.f, 0.f, 0.f, 0.f, 0.f, 0.f, 0.f};
#pragma unroll
    for (int n = 0; n < 5; ++n) acc[n] = vzero;

    for (int c = 0; c < NCHUNKS; ++c) {
        // ---- build bf16 feature chunk [16 x 768] in LDS ----
        for (int r = 0; r < ROWS; ++r) {
            const float* gim = g_i + (size_t)mi_s[r] * GDIM;
            const float* gia = g_i + (size_t)ai_s[r] * GDIM;
            for (int col = t; col < CHUNK; col += 64) {
                int k = c * CHUNK + col;
                float v;
                if (k < GDIM)                v = gim[k];
                else if (k < 2 * GDIM)       v = gia[k - GDIM];
                else if (k < 3 * GDIM)       v = gim[k - 2 * GDIM] * gia[k - 2 * GDIM];
                else if (k < 3 * GDIM + 20)  v = dist_emb[di_s[r] * 20 + (k - 3000)];
                else if (k < 3 * GDIM + 40)  v = genre_emb[ge_s[r] * 20 + (k - 3020)];
                else if (k < GIJ_K)          v = speaker_emb[sp_s[r] * 20 + (k - 3040)];
                else                         v = 0.0f;
                Atile[r][col] = (__bf16)v;
            }
        }
        __syncthreads();

        // ---- GEMM1 over this chunk ----
        const char* arow = (const char*)&Atile[mrow][0];
        for (int ks = 0; ks < KS_PER_CHUNK; ++ks) {
            Frag a;
            a.u[0] = *(const uint4*)(arow + ks * 64 + half * 16);
            a.u[1] = *(const uint4*)(arow + ks * 64 + half * 16 + 32);
            int ksg = c * KS_PER_CHUNK + ks;
#pragma unroll
            for (int n = 0; n < 5; ++n) {
                int ntg = wave * 5 + n;
                const uint4* bp =
                    (const uint4*)(W1b + (((size_t)ksg * NT1 + ntg) * 32 + lane) * 16);
                Frag b;
                b.u[0] = bp[0];
                b.u[1] = bp[1];
                acc[n] = __builtin_amdgcn_wmma_f32_16x16x32_bf16(
                    false, a.v, false, b.v, (short)0, acc[n], false, false);
            }
        }
        __syncthreads();   // protect Atile before next chunk rewrite
    }

    // ---- bias1 + ReLU -> bf16 h-tile (C layout: row = half*8+r, col = lane&15) ----
#pragma unroll
    for (int n = 0; n < 5; ++n) {
        int col  = (wave * 5 + n) * 16 + mrow;
        float bs = (col < HDIM) ? b1[col] : 0.0f;
#pragma unroll
        for (int r = 0; r < 8; ++r) {
            float h = acc[n][r] + bs;
            h = h > 0.0f ? h : 0.0f;
            Htile[half * 8 + r][col] = (__bf16)h;
        }
    }
    __syncthreads();

    // ---- GEMM2: h (16x160) @ W2b (160x160) ----
    v8f acc2[5];
#pragma unroll
    for (int n = 0; n < 5; ++n) acc2[n] = vzero;

    const char* hrow = (const char*)&Htile[mrow][0];
#pragma unroll
    for (int ks = 0; ks < KS2; ++ks) {
        Frag a;
        a.u[0] = *(const uint4*)(hrow + ks * 64 + half * 16);
        a.u[1] = *(const uint4*)(hrow + ks * 64 + half * 16 + 32);
#pragma unroll
        for (int n = 0; n < 5; ++n) {
            int ntg = wave * 5 + n;
            const uint4* bp =
                (const uint4*)(W2b + (((size_t)ks * NT1 + ntg) * 32 + lane) * 16);
            Frag b;
            b.u[0] = bp[0];
            b.u[1] = bp[1];
            acc2[n] = __builtin_amdgcn_wmma_f32_16x16x32_bf16(
                false, a.v, false, b.v, (short)0, acc2[n], false, false);
        }
    }

    // ---- bias2 + ReLU + dot with W3, reduced across lanes ----
    float partial[8];
#pragma unroll
    for (int r = 0; r < 8; ++r) partial[r] = 0.0f;
#pragma unroll
    for (int n = 0; n < 5; ++n) {
        int col  = (wave * 5 + n) * 16 + mrow;
        float bs = (col < HDIM) ? b2[col] : 0.0f;
        float w3 = (col < HDIM) ? W3[col] : 0.0f;
#pragma unroll
        for (int r = 0; r < 8; ++r) {
            float h = acc2[n][r] + bs;
            h = h > 0.0f ? h : 0.0f;
            partial[r] += h * w3;
        }
    }
#pragma unroll
    for (int r = 0; r < 8; ++r) {
#pragma unroll
        for (int o = 8; o; o >>= 1)
            partial[r] += __shfl_xor(partial[r], o, 16);  // within 16-lane N groups
    }
    if (mrow == 0) {
#pragma unroll
        for (int r = 0; r < 8; ++r) red[half * 8 + r][wave] = partial[r];
    }
    __syncthreads();

    if (t < ROWS) {
        int p = p0 + t;
        if (p < P) {
            float sij = red[t][0] + red[t][1] + b3[0];
            scores[p] = sij + mention_scores[mi_s[t]] + mention_scores[ai_s[t]];
        }
    }
}

// ---------------------------------------------------------------------------
// Per-group softmax over [scores, epsilon=0] and grid fill. 1 wave per group.
// ---------------------------------------------------------------------------
__global__ __launch_bounds__(32)
void softmax_grid(const float* __restrict__ scores,
                  const int* __restrict__ lengths,
                  const int* __restrict__ starts,
                  float* __restrict__ grid) {
    int g     = blockIdx.x;
    int lane  = threadIdx.x;
    int start = starts[g];
    int len   = lengths[g];

    float mx = -3.402823466e38f;
    for (int k = lane; k < len; k += 32) mx = fmaxf(mx, scores[start + k]);
#pragma unroll
    for (int o = 16; o; o >>= 1) mx = fmaxf(mx, __shfl_xor(mx, o, 32));
    mx = fmaxf(mx, 0.0f);   // epsilon logit = 0 participates in the max

    float s = 0.0f;
    for (int k = lane; k < len; k += 32) s += expf(scores[start + k] - mx);
#pragma unroll
    for (int o = 16; o; o >>= 1) s += __shfl_xor(s, o, 32);

    float eps_e = expf(-mx);
    float inv   = 1.0f / (s + eps_e);

    float* row = grid + (size_t)(g + 1) * KMAXP1;
    for (int k = lane; k < len; k += 32)
        row[k] = expf(scores[start + k] - mx) * inv;
    if (lane == 0) row[len] = eps_e * inv;
}

// ---------------------------------------------------------------------------
// Launch
// ---------------------------------------------------------------------------
extern "C" void kernel_launch(void* const* d_in, const int* in_sizes, int n_in,
                              void* d_out, int out_size, void* d_ws, size_t ws_size,
                              hipStream_t stream) {
    const float* g_i            = (const float*)d_in[0];
    const float* mention_scores = (const float*)d_in[1];
    const float* dist_emb       = (const float*)d_in[2];
    const float* genre_emb      = (const float*)d_in[3];
    const float* speaker_emb    = (const float*)d_in[4];
    const float* W1             = (const float*)d_in[5];
    const float* b1             = (const float*)d_in[6];
    const float* W2             = (const float*)d_in[7];
    const float* b2             = (const float*)d_in[8];
    const float* W3             = (const float*)d_in[9];
    const float* b3             = (const float*)d_in[10];
    const int* mention_ids    = (const int*)d_in[11];
    const int* antecedent_ids = (const int*)d_in[12];
    const int* dist_ids       = (const int*)d_in[13];
    const int* genre_ids      = (const int*)d_in[14];
    const int* speaker_ids    = (const int*)d_in[15];
    const int* lengths        = (const int*)d_in[16];
    // d_in[17] seg_ids / d_in[18] offsets: recomputed via starts scan

    const int P = in_sizes[11];
    const int M = in_sizes[16];

    // Workspace layout (all 256B-aligned):
    //   W1b: KS1*NT1*512 bf16 = 983,040 B
    //   W2b: KS2*NT1*512 bf16 =  51,200 B
    //   scores: P floats
    //   starts: M ints
    char* ws = (char*)d_ws;
    const size_t w1b_bytes = (size_t)KS1 * NT1 * 512 * sizeof(__bf16);
    const size_t w2b_bytes = (size_t)KS2 * NT1 * 512 * sizeof(__bf16);
    const size_t sc_bytes  = (((size_t)P * 4) + 255) & ~(size_t)255;
    __bf16* W1b   = (__bf16*)ws;
    __bf16* W2b   = (__bf16*)(ws + w1b_bytes);
    float*  sc    = (float*)(ws + w1b_bytes + w2b_bytes);
    int*    starts= (int*)(ws + w1b_bytes + w2b_bytes + sc_bytes);

    // 1) pack weights to bf16 fragment layout
    {
        int total1 = KS1 * NT1 * 512;
        pack_w_bf16<<<(total1 + 255) / 256, 256, 0, stream>>>(W1, W1b, GIJ_K, HDIM, total1);
        int total2 = KS2 * NT1 * 512;
        pack_w_bf16<<<(total2 + 255) / 256, 256, 0, stream>>>(W2, W2b, HDIM, HDIM, total2);
    }
    // 2) init output grid
    init_grid<<<(out_size + 255) / 256, 256, 0, stream>>>((float*)d_out, out_size);
    // 3) group start offsets
    scan_starts<<<1, 256, 0, stream>>>(lengths, starts, M);
    // 4) fused gather + 3-layer MLP (WMMA) -> pair scores
    pair_mlp<<<(P + ROWS - 1) / ROWS, 64, 0, stream>>>(
        g_i, mention_scores, dist_emb, genre_emb, speaker_emb,
        W1b, b1, W2b, b2, W3, b3,
        mention_ids, antecedent_ids, dist_ids, genre_ids, speaker_ids,
        sc, P);
    // 5) ragged softmax (+epsilon) into the padded grid
    softmax_grid<<<M, 32, 0, stream>>>(sc, lengths, starts, (float*)d_out);
}
